// CrossAttention_28183575396415
// MI455X (gfx1250) — matrix operand order repaired and checked by reading, not compiled, not run on hardware
//
#include <hip/hip_runtime.h>
#include <hip/hip_bf16.h>

// ---------- problem constants (from reference) ----------
#define C_DIM 1024
#define TKV   2048
#define BATCH 4
#define LREP  3
#define ROWS  (BATCH * TKV)   // 8192 rows of x_kv (flattened)
#define TQ    (TKV * LREP)    // 6144

// ---------- vector types ----------
typedef __attribute__((ext_vector_type(16))) __bf16          v16bf;
typedef __attribute__((ext_vector_type(8)))  __bf16          v8bf;
typedef __attribute__((ext_vector_type(8)))  float           v8f;
typedef __attribute__((ext_vector_type(8)))  unsigned short  v8us;
typedef __attribute__((ext_vector_type(4)))  unsigned int    v4u;
typedef __attribute__((ext_vector_type(4)))  int             v4i;
typedef __attribute__((ext_vector_type(8)))  int             v8i;

#if __has_builtin(__builtin_amdgcn_tensor_load_to_lds)
#define HAVE_TDM 1
#else
#define HAVE_TDM 0
#endif

// ---------- fp32 -> bf16 (round-to-nearest-even) ----------
static __device__ __forceinline__ unsigned short f2bf_rne(float f) {
    unsigned int u = __float_as_uint(f);
    unsigned int r = (u + 0x7FFFu + ((u >> 16) & 1u)) >> 16;
    return (unsigned short)r;
}

// ---------- WMMA fragment helpers ----------
// A fragment (16x32 A, k order chosen here, matched by B): lane {m=lane&15, h=lane>>4} holds
// K = {k0+h*16 .. +15} contiguously in elements 0..15.
static __device__ __forceinline__ v16bf load_a_frag(const unsigned short* row, int k0, int lh) {
    v8bf alo = *(const v8bf*)(row + k0 + lh * 16);
    v8bf ahi = *(const v8bf*)(row + k0 + lh * 16 + 8);
    v16bf a;
    #pragma unroll
    for (int i = 0; i < 8; ++i) { a[i] = alo[i]; a[8 + i] = ahi[i]; }
    return a;
}

// ---------- kernel 1: straight fp32 -> bf16 conversion ----------
__global__ void __launch_bounds__(256)
cvt_f32_bf16_x4(const float* __restrict__ in, unsigned short* __restrict__ out) {
    size_t i = ((size_t)blockIdx.x * blockDim.x + threadIdx.x) * 4;
    float4 v = *(const float4*)(in + i);
    ushort4 o;
    o.x = f2bf_rne(v.x);
    o.y = f2bf_rne(v.y);
    o.z = f2bf_rne(v.z);
    o.w = f2bf_rne(v.w);
    *(ushort4*)(out + i) = o;
}

// ---------- kernel 2: transpose + convert (Wv[c][m] -> WvT16[m][c]) ----------
__global__ void __launch_bounds__(256)
transpose_cvt_bf16(const float* __restrict__ in, unsigned short* __restrict__ out, int n) {
    __shared__ unsigned short tile[32][33];
    int x  = blockIdx.x * 32 + threadIdx.x;
    int y0 = blockIdx.y * 32;
    #pragma unroll
    for (int j = threadIdx.y; j < 32; j += 8)
        tile[j][threadIdx.x] = f2bf_rne(in[(size_t)(y0 + j) * n + x]);
    __syncthreads();
    int xo  = y0 + threadIdx.x;
    int yo0 = blockIdx.x * 32;
    #pragma unroll
    for (int j = threadIdx.y; j < 32; j += 8)
        out[(size_t)(yo0 + j) * n + xo] = tile[threadIdx.x][j];
}

// ---------- kernel 3: WcT = Wproj @ Wv  (bf16 in, bf16 out) ----------
// A  = Wp16  [M=1024][K=1024] row-major; Bt = WvT16 [N=1024][K=1024] row-major.
// Cout = WcT[j][m] bf16 row-major (serves as Bt of kernel 4).
__global__ void __launch_bounds__(256)
gemm_bf16_bf16out(const unsigned short* __restrict__ A,
                  const unsigned short* __restrict__ Bt,
                  unsigned short* __restrict__ Cout) {
    const int K = C_DIM;
    const int lane = threadIdx.x & 31;
    const int wid  = threadIdx.x >> 5;
    const int lm = lane & 15;
    const int lh = lane >> 4;

    const int m0 = blockIdx.y * 32  + (wid & 1) * 16;
    const int n0 = blockIdx.x * 256 + (wid >> 1) * 64;

    v8f acc0 = {}, acc1 = {}, acc2 = {}, acc3 = {};

    const unsigned short* arow = A + (size_t)(m0 + lm) * K;
    const unsigned short* br0  = Bt + (size_t)(n0 +  0 + lm) * K + lh * 16;
    const unsigned short* br1  = Bt + (size_t)(n0 + 16 + lm) * K + lh * 16;
    const unsigned short* br2  = Bt + (size_t)(n0 + 32 + lm) * K + lh * 16;
    const unsigned short* br3  = Bt + (size_t)(n0 + 48 + lm) * K + lh * 16;

    for (int k0 = 0; k0 < K; k0 += 32) {
        v16bf a  = load_a_frag(arow, k0, lh);
        v16bf b0 = *(const v16bf*)(br0 + k0);
        v16bf b1 = *(const v16bf*)(br1 + k0);
        v16bf b2 = *(const v16bf*)(br2 + k0);
        v16bf b3 = *(const v16bf*)(br3 + k0);

        acc0 = __builtin_amdgcn_wmma_f32_16x16x32_bf16(false, a, false, b0, (short)0, acc0, false, false);
        acc1 = __builtin_amdgcn_wmma_f32_16x16x32_bf16(false, a, false, b1, (short)0, acc1, false, false);
        acc2 = __builtin_amdgcn_wmma_f32_16x16x32_bf16(false, a, false, b2, (short)0, acc2, false, false);
        acc3 = __builtin_amdgcn_wmma_f32_16x16x32_bf16(false, a, false, b3, (short)0, acc3, false, false);
    }

    #pragma unroll
    for (int i = 0; i < 8; ++i) {
        size_t row = (size_t)(m0 + lh * 8 + i) * C_DIM;
        Cout[row + n0 +  0 + lm] = f2bf_rne(acc0[i]);
        Cout[row + n0 + 16 + lm] = f2bf_rne(acc1[i]);
        Cout[row + n0 + 32 + lm] = f2bf_rne(acc2[i]);
        Cout[row + n0 + 48 + lm] = f2bf_rne(acc3[i]);
    }
}

// ---------- TDM staging of the B panel into LDS ----------
// Loads a 2-D tile: 256 rows (N) x 32 cols (K) of bf16 (16 KB) from Bt into LDS.
static __device__ __forceinline__ void
stage_B(unsigned short* dst, const unsigned short* __restrict__ Bt,
        int nbase, int k0, int wid) {
#if HAVE_TDM
    if (wid == 0) {
        unsigned long long ga =
            (unsigned long long)(uintptr_t)(Bt + (size_t)nbase * C_DIM + k0);
        unsigned lds = (unsigned)(uintptr_t)dst;  // low 32 bits = LDS byte address
        // D# group 0: count=1 | lds_addr | global_addr[56:0] | type=2
        v4u g0 = { 1u,
                   lds,
                   (unsigned)(ga & 0xFFFFFFFFu),
                   ((unsigned)((ga >> 32) & 0x01FFFFFFu)) | 0x80000000u };
        // D# group 1: data_size=2B; tensor_dim0=1024; tensor_dim1=1024;
        // tile_dim0=32 (K slice); tile_dim1=256 (rows); tensor_dim0_stride=1024.
        v8i g1 = { (int)0x00010000u,        // wg_mask=0, data_size=1 (2 bytes)
                   (int)(C_DIM << 16),      // [31:16] = tensor_dim0[15:0]
                   (int)(C_DIM << 16),      // [15:0]=td0 hi (0), [31:16]=tensor_dim1[15:0]
                   (int)(32 << 16),         // [15:0]=td1 hi (0), [31:16]=tile_dim0=32
                   (int)256,                // [15:0]=tile_dim1=256, [31:16]=tile_dim2=0
                   (int)C_DIM,              // tensor_dim0_stride[31:0]=1024
                   0,                       // stride hi | tensor_dim1_stride lo
                   0 };
        v4i z4 = { 0, 0, 0, 0 };
#if defined(__clang_major__) && (__clang_major__ >= 23)
        v8i z8 = { 0, 0, 0, 0, 0, 0, 0, 0 };
        __builtin_amdgcn_tensor_load_to_lds(g0, g1, z4, z4, z8, 0);
#else
        __builtin_amdgcn_tensor_load_to_lds(g0, g1, z4, z4, 0);
#endif
    }
#else
    // Fallback: cooperative synchronous copy (all 256 threads, 16B chunks).
    const unsigned short* src = Bt + (size_t)nbase * C_DIM + k0;
    for (int idx = threadIdx.x; idx < 256 * 4; idx += 256) {
        int r = idx >> 2;
        int c = (idx & 3) * 8;
        *(v8us*)(dst + r * 32 + c) = *(const v8us*)(src + (size_t)r * C_DIM + c);
    }
#endif
}

static __device__ __forceinline__ void stage_commit(int wid) {
#if HAVE_TDM
#if __has_builtin(__builtin_amdgcn_s_wait_tensorcnt)
    if (wid == 0) __builtin_amdgcn_s_wait_tensorcnt(0);
#endif
#endif
    (void)wid;
    __syncthreads();
}

// ---------- kernel 4: Out = x_kv @ Wcomb, fused 3x row replication ----------
// A  = X16 [8192][1024] bf16 (pre-converted x_kv), Bt = WcT16 [N][K] bf16.
// WG = 64 M x 256 N; 8 waves as 2(M) x 4(N); wave tile = 32 M x 64 N (8 WMMA/k).
// B panel double-buffered through LDS via TDM.
__global__ void __launch_bounds__(256)
gemm_xw_fused(const unsigned short* __restrict__ X16,
              const unsigned short* __restrict__ Bt,
              float* __restrict__ Out) {
    __shared__ unsigned short bsmem[2][256 * 32];   // 2 x 16 KB

    const int K = C_DIM;
    const int lane = threadIdx.x & 31;
    const int wid  = threadIdx.x >> 5;
    const int lm = lane & 15;
    const int lh = lane >> 4;
    const int mwave = wid >> 2;                     // 0..1
    const int nwave = wid & 3;                      // 0..3

    const int m0    = blockIdx.y * 64 + mwave * 32;
    const int nbase = blockIdx.x * 256;

    v8f acc[2][4] = {};

    const unsigned short* arow0 = X16 + (size_t)(m0 + lm) * K;
    const unsigned short* arow1 = X16 + (size_t)(m0 + 16 + lm) * K;

    stage_B(&bsmem[0][0], Bt, nbase, 0, wid);
    stage_commit(wid);

    for (int k0 = 0; k0 < K; k0 += 32) {
        const int cur = (k0 >> 5) & 1;
        const bool more = (k0 + 32) < K;
        if (more) stage_B(&bsmem[cur ^ 1][0], Bt, nbase, k0 + 32, wid);

        __builtin_prefetch(arow0 + k0 + 64, 0, 0);  // next A block -> global_prefetch_b8

        v16bf a0 = load_a_frag(arow0, k0, lh);
        v16bf a1 = load_a_frag(arow1, k0, lh);

        // B fragments from LDS: panel row r = nwave*64 + nt*16 + lm, 16 bf16 at K=lh*16
        const unsigned short* bb = &bsmem[cur][0] + ((nwave * 64 + lm) * 32 + lh * 16);
        v16bf b0 = *(const v16bf*)(bb);
        v16bf b1 = *(const v16bf*)(bb + 16 * 32);
        v16bf b2 = *(const v16bf*)(bb + 32 * 32);
        v16bf b3 = *(const v16bf*)(bb + 48 * 32);

        acc[0][0] = __builtin_amdgcn_wmma_f32_16x16x32_bf16(false, a0, false, b0, (short)0, acc[0][0], false, false);
        acc[0][1] = __builtin_amdgcn_wmma_f32_16x16x32_bf16(false, a0, false, b1, (short)0, acc[0][1], false, false);
        acc[0][2] = __builtin_amdgcn_wmma_f32_16x16x32_bf16(false, a0, false, b2, (short)0, acc[0][2], false, false);
        acc[0][3] = __builtin_amdgcn_wmma_f32_16x16x32_bf16(false, a0, false, b3, (short)0, acc[0][3], false, false);
        acc[1][0] = __builtin_amdgcn_wmma_f32_16x16x32_bf16(false, a1, false, b0, (short)0, acc[1][0], false, false);
        acc[1][1] = __builtin_amdgcn_wmma_f32_16x16x32_bf16(false, a1, false, b1, (short)0, acc[1][1], false, false);
        acc[1][2] = __builtin_amdgcn_wmma_f32_16x16x32_bf16(false, a1, false, b2, (short)0, acc[1][2], false, false);
        acc[1][3] = __builtin_amdgcn_wmma_f32_16x16x32_bf16(false, a1, false, b3, (short)0, acc[1][3], false, false);

        if (more) stage_commit(wid);
    }

    // Epilogue: C layout (VGPR i: lanes 0-15 -> M=i, lanes 16-31 -> M=8+i; N=lane&15)
    #pragma unroll
    for (int mt = 0; mt < 2; ++mt) {
        #pragma unroll
        for (int i = 0; i < 8; ++i) {
            int srow = m0 + mt * 16 + lh * 8 + i;   // row in flattened (B,Tkv)
            int b = srow >> 11;                     // / 2048
            int s = srow & 2047;
            float* orow = Out + ((size_t)b * TQ + (size_t)s * LREP) * C_DIM;
            int col0 = nbase + nwave * 64 + lm;
            float v0 = acc[mt][0][i], v1 = acc[mt][1][i];
            float v2 = acc[mt][2][i], v3 = acc[mt][3][i];
            #pragma unroll
            for (int l = 0; l < LREP; ++l) {
                float* p = orow + (size_t)l * C_DIM;
                p[col0]      = v0;
                p[col0 + 16] = v1;
                p[col0 + 32] = v2;
                p[col0 + 48] = v3;
            }
        }
    }
}

extern "C" void kernel_launch(void* const* d_in, const int* in_sizes, int n_in,
                              void* d_out, int out_size, void* d_ws, size_t ws_size,
                              hipStream_t stream) {
    // inputs: 0=x_q (unused), 1=x_kv, 2=Wq (unused), 3=Wk (unused), 4=Wv, 5=Wproj
    const float* x_kv  = (const float*)d_in[1];
    const float* Wv    = (const float*)d_in[4];
    const float* Wproj = (const float*)d_in[5];
    float* out = (float*)d_out;

    // workspace layout (bf16 stored as ushort)
    unsigned short* Wp16  = (unsigned short*)d_ws;               // 2 MB: Wproj bf16 [j][c]
    unsigned short* WvT16 = Wp16  + (size_t)C_DIM * C_DIM;       // 2 MB: Wv^T bf16  [m][c]
    unsigned short* WcT16 = WvT16 + (size_t)C_DIM * C_DIM;       // 2 MB: (Wproj@Wv) [j][m]
    unsigned short* X16   = WcT16 + (size_t)C_DIM * C_DIM;       // 16 MB: x_kv bf16 [s][m]

    // 1) convert Wproj and x_kv to bf16
    cvt_f32_bf16_x4<<<(C_DIM * C_DIM) / (256 * 4), 256, 0, stream>>>(Wproj, Wp16);
    cvt_f32_bf16_x4<<<((size_t)ROWS * C_DIM) / (256 * 4), 256, 0, stream>>>(x_kv, X16);

    // 2) transpose-convert Wv -> WvT16[m][c]
    transpose_cvt_bf16<<<dim3(C_DIM / 32, C_DIM / 32), dim3(32, 8), 0, stream>>>(Wv, WvT16, C_DIM);

    // 3) WcT[j][m] = sum_c Wproj[j][c] * Wv[c][m]
    gemm_bf16_bf16out<<<dim3(C_DIM / 256, C_DIM / 32), 256, 0, stream>>>(Wp16, WvT16, WcT16);

    // 4) Out[b][s*3+l][j] = sum_m x_kv[b][s][m] * WcT[j][m]  (TDM-staged B panel)
    gemm_xw_fused<<<dim3(C_DIM / 256, ROWS / 64), 256, 0, stream>>>(X16, WcT16, out);
}
